// MultiQueryAttention_56959856279849
// MI455X (gfx1250) — compile-verified, matrix operand order checked by reading, not run on hardware
//
#include <hip/hip_runtime.h>
#include <hip/hip_bf16.h>

// ---------------------------------------------------------------------------
// MQA forward for MI455X (gfx1250, wave32, WMMA + Tensor Data Mover).
// f32->bf16 converts -> WMMA GEMM projections -> flash attention with
// TDM-staged K/V chunks in LDS (double buffered, s_wait_tensorcnt) ->
// WMMA GEMM output projection (f32).
// ---------------------------------------------------------------------------

typedef __bf16 bf16;
typedef __attribute__((ext_vector_type(16))) __bf16 v16bf;
typedef __attribute__((ext_vector_type(8)))  __bf16 v8bf;
typedef __attribute__((ext_vector_type(8)))  float  v8f;
typedef __attribute__((ext_vector_type(4)))  unsigned int u32x4;
typedef __attribute__((ext_vector_type(8)))  int i32x8;
typedef __attribute__((ext_vector_type(4)))  int i32x4;

#if defined(__has_builtin)
#if __has_builtin(__builtin_amdgcn_tensor_load_to_lds) && \
    __has_builtin(__builtin_amdgcn_s_wait_tensorcnt)
#define HAVE_TDM 1
#endif
#endif
#ifndef HAVE_TDM
#define HAVE_TDM 0
#endif

// Compile-time probes (visible in hipcc stderr) so we can see which path built.
#define CDNA5_STR2(x) #x
#define CDNA5_STR(x) CDNA5_STR2(x)
#if HAVE_TDM
#pragma message("CDNA5 probe: HAVE_TDM=1 (tensor_load_to_lds path), clang=" CDNA5_STR(__clang_major__))
#else
#pragma message("CDNA5 probe: HAVE_TDM=0 (cooperative-copy fallback), clang=" CDNA5_STR(__clang_major__))
#endif

__device__ __forceinline__ bf16 f2bf(float f) {
  union { float f; unsigned u; } in;
  in.f = f;
  unsigned u = in.u;
  unsigned r = u + 0x7FFFu + ((u >> 16) & 1u);   // round-to-nearest-even
  union { unsigned short s; bf16 b; } out;
  out.s = (unsigned short)(r >> 16);
  return out.b;
}

// Load one 16x32 bf16 A/B fragment from a row-major matrix (global or LDS).
// ISA 16-bit A 16x32 layout: lanes 0-15: row=lane, K={k0..k0+7, k0+16..k0+23};
// lanes 16-31: row=lane-16, K={k0+8..k0+15, k0+24..k0+31}. B mirrored (lane=N).
__device__ __forceinline__ v16bf load_frag16(const bf16* base, long ld, int lane) {
  const int r = lane & 15, half = lane >> 4;
  union { v16bf v; v8bf h[2]; } f;
  const bf16* p = base + (long)r * ld + half * 8;
  f.h[0] = *(const v8bf*)(p);
  f.h[1] = *(const v8bf*)(p + 16);
  return f.v;
}

#if HAVE_TDM
__device__ __forceinline__ unsigned lds_offset(const void* p) {
  // LDS (addrspace 3) addresses are byte offsets into the workgroup's LDS.
  return (unsigned)(unsigned long long)(const __attribute__((address_space(3))) void*)p;
}

// Issue a TDM 2D tile load (bf16 elements) into LDS.  D# packing per ISA
// ch.8: group0 = {count, lds_addr, global_addr[56:0], type=2}; group1 carries
// data_size=2B, tensor_dim0/1, tile_dim0/1, tensor_dim0_stride.
__device__ __forceinline__ void tdm_load_2d(const bf16* gptr, unsigned lds_off,
                                            unsigned ten_d0, unsigned ten_d1,
                                            unsigned long long d0_stride,
                                            unsigned til_d0, unsigned til_d1) {
  unsigned long long ga = (unsigned long long)gptr;
  u32x4 g0;
  g0[0] = 1u;                                             // count=1 (valid D#)
  g0[1] = lds_off;                                        // lds_addr (bytes)
  g0[2] = (unsigned)(ga & 0xffffffffu);                   // global_addr[31:0]
  g0[3] = (unsigned)((ga >> 32) & 0x1ffffffu) | (2u << 30); // [56:32] | type=2
  i32x8 g1;
  g1[0] = (int)(1u << 16);                                // data_size=1 (2B)
  g1[1] = (int)((ten_d0 & 0xffffu) << 16);                // tensor_dim0[15:0]
  g1[2] = (int)(((ten_d0 >> 16) & 0xffffu) | ((ten_d1 & 0xffffu) << 16));
  g1[3] = (int)(((ten_d1 >> 16) & 0xffffu) | ((til_d0 & 0xffffu) << 16));
  g1[4] = (int)(til_d1 & 0xffffu);                        // tile_dim1, dim2=0
  g1[5] = (int)(unsigned)(d0_stride & 0xffffffffu);       // dim0_stride[31:0]
  g1[6] = (int)(unsigned)((d0_stride >> 32) & 0xffffu);   // dim0_stride[47:32]
  g1[7] = 0;
  i32x4 z4 = {0, 0, 0, 0};
#if defined(__clang_major__) && (__clang_major__ >= 23)
  i32x8 z8 = {0, 0, 0, 0, 0, 0, 0, 0};
  __builtin_amdgcn_tensor_load_to_lds(g0, g1, z4, z4, z8, 0);
#else
  __builtin_amdgcn_tensor_load_to_lds(g0, g1, z4, z4, 0);
#endif
}
#endif  // HAVE_TDM

// ---------------------------------------------------------------------------
__global__ void f32_to_bf16_kernel(const float* __restrict__ src,
                                   bf16* __restrict__ dst, long n) {
  long i = (long)blockIdx.x * blockDim.x + threadIdx.x;
  long stride = (long)gridDim.x * blockDim.x;
  for (; i < n; i += stride) dst[i] = f2bf(src[i]);
}

// ---------------------------------------------------------------------------
// C[M,N] = A[M,K](bf16) @ W[N,K]^T(bf16) + bias; wave tile 16x64 (4 C frags).
// K loop unrolled by 2 with two independent fragment sets (even/odd stage):
// every stage refill is a fresh load into the same virtual registers, so the
// pipeline needs no register-rotation moves (K is a multiple of 64 here).
// mode 0: f32 row-major; 1: bf16 row-major; 2: bf16 [B,H,S,64] head-permute;
// mode 3: bf16 transposed [B,64,S] (for V so PV B-fragments load contiguously).
__global__ __launch_bounds__(256) void gemm_bt_kernel(
    const bf16* __restrict__ A, const bf16* __restrict__ W,
    const float* __restrict__ bias, float* __restrict__ outF,
    bf16* __restrict__ outB, int M, int N, int K, int mode) {
  const int lane = threadIdx.x & 31;
  const int wave = threadIdx.x >> 5;
  const int m0 = (blockIdx.x * 8 + wave) * 16;
  const int n0 = blockIdx.y * 64;
  if (m0 >= M) return;  // wave-uniform

  v8f acc[4];
#pragma unroll
  for (int t = 0; t < 4; ++t)
#pragma unroll
    for (int i = 0; i < 8; ++i) acc[t][i] = 0.f;

  const bf16* Abase = A + (long)m0 * K;
  const bf16* Wbase = W + (long)n0 * K;

  // Even-stage fragments for k = 0.
  v16bf a0 = load_frag16(Abase, K, lane);
  v16bf w0[4];
#pragma unroll
  for (int t = 0; t < 4; ++t) w0[t] = load_frag16(Wbase + (long)t * 16 * K, K, lane);

  for (int k0 = 0; k0 < K; k0 += 64) {
    // Fetch odd stage (k0+32) while even stage computes.
    __builtin_prefetch(Abase + k0 + 64, 0, 3);   // global_prefetch_b8
    v16bf a1 = load_frag16(Abase + k0 + 32, K, lane);
    v16bf w1[4];
#pragma unroll
    for (int t = 0; t < 4; ++t)
      w1[t] = load_frag16(Wbase + (long)t * 16 * K + k0 + 32, K, lane);

#pragma unroll
    for (int t = 0; t < 4; ++t)
      acc[t] = __builtin_amdgcn_wmma_f32_16x16x32_bf16(
          false, a0, false, w0[t], (short)0, acc[t], false, false);

    // Refill even stage (k0+64) while odd stage computes.
    if (k0 + 64 < K) {
      a0 = load_frag16(Abase + k0 + 64, K, lane);
#pragma unroll
      for (int t = 0; t < 4; ++t)
        w0[t] = load_frag16(Wbase + (long)t * 16 * K + k0 + 64, K, lane);
    }

#pragma unroll
    for (int t = 0; t < 4; ++t)
      acc[t] = __builtin_amdgcn_wmma_f32_16x16x32_bf16(
          false, a1, false, w1[t], (short)0, acc[t], false, false);
  }

  // C layout: VGPR i -> row = i + 8*(lane>>4), col = n0 + t*16 + (lane&15)
  const int r = lane & 15, half = lane >> 4;
#pragma unroll
  for (int t = 0; t < 4; ++t) {
    const int col = n0 + t * 16 + r;
    const float bv = bias[col];
#pragma unroll
    for (int i = 0; i < 8; ++i) {
      const int row = m0 + i + 8 * half;
      const float v = acc[t][i] + bv;
      if (mode == 0) {
        outF[(long)row * N + col] = v;
      } else if (mode == 1) {
        outB[(long)row * N + col] = f2bf(v);
      } else if (mode == 2) {
        const long b = row >> 11, s = row & 2047;   // S = 2048
        const long h = col >> 6, d = col & 63;      // d_k = 64
        outB[((b * 16 + h) * 2048 + s) * 64 + d] = f2bf(v);
      } else {                                      // mode 3: V^T
        const long b = row >> 11, s = row & 2047;
        outB[((long)b * 64 + col) * 2048 + s] = f2bf(v);
      }
    }
  }
}

// ---------------------------------------------------------------------------
// Flash MQA: 4 waves/block = 4 consecutive 16-row query tiles of one (b,h);
// all share the same K/V stream, so K (32x64) and V^T (64x32) chunks are
// DMA'd into double-buffered LDS by the Tensor Data Mover (waves 0/1 issue,
// s_wait_tensorcnt + barrier synchronizes), overlapping with WMMA + softmax.
// Qp [B,H,S,64], Kp [B,S,64], VTp [B,64,S], Ctx [B,S,1024].
__global__ __launch_bounds__(128) void mqa_flash_kernel(
    const bf16* __restrict__ Qp, const bf16* __restrict__ Kp,
    const bf16* __restrict__ VTp, bf16* __restrict__ Ctx) {
  __shared__ __attribute__((aligned(16))) bf16 Kch[2][32 * 64];
  __shared__ __attribute__((aligned(16))) bf16 Vch[2][64 * 32];
  __shared__ __attribute__((aligned(16))) bf16 Plds[4][16 * 32];

  const int lane = threadIdx.x & 31;
  const int wave = threadIdx.x >> 5;
  const int gw = blockIdx.x * 4 + wave;
  const int tile = gw & 127;           // S/16 = 128 query tiles per head
  const int h = (gw >> 7) & 15;
  const int b = gw >> 11;
  const int m0 = tile * 16;
  const int r = lane & 15, half = lane >> 4;

  const bf16* qbase = Qp + (((long)b * 16 + h) * 2048 + m0) * 64;
  const v16bf qa0 = load_frag16(qbase + 0, 64, lane);   // d_k 0..31
  const v16bf qa1 = load_frag16(qbase + 32, 64, lane);  // d_k 32..63

  const bf16* kbatch = Kp + (long)b * 2048 * 64;
  const bf16* vbatch = VTp + (long)b * 64 * 2048;

  float rmax[8], rsum[8];
  v8f oacc[4];
#pragma unroll
  for (int i = 0; i < 8; ++i) { rmax[i] = -1e30f; rsum[i] = 0.f; }
#pragma unroll
  for (int t = 0; t < 4; ++t)
#pragma unroll
    for (int i = 0; i < 8; ++i) oacc[t][i] = 0.f;

  bf16* pl = &Plds[wave][0];

#if HAVE_TDM
  // Prologue: DMA chunk 0 into buffer 0.
  if (wave == 0)
    tdm_load_2d(kbatch, lds_offset(&Kch[0][0]), 64, 4096, 64, 64, 32);
  if (wave == 1)
    tdm_load_2d(vbatch, lds_offset(&Vch[0][0]), 2048, 128, 2048, 32, 64);
  __builtin_amdgcn_s_wait_tensorcnt(0);
  __syncthreads();
#endif

  for (int j0 = 0; j0 < 2048; j0 += 32) {
    const int cur = (j0 >> 5) & 1;
#if HAVE_TDM
    // Kick off DMA of the next chunk into the other buffer (overlaps compute).
    if (j0 + 32 < 2048) {
      const int nxt = cur ^ 1;
      if (wave == 0)
        tdm_load_2d(kbatch + (long)(j0 + 32) * 64, lds_offset(&Kch[nxt][0]),
                    64, 4096, 64, 64, 32);
      if (wave == 1)
        tdm_load_2d(vbatch + (j0 + 32), lds_offset(&Vch[nxt][0]),
                    2048, 128, 2048, 32, 64);
    }
#else
    {  // Fallback: cooperative synchronous staging.
      const int tid = threadIdx.x;
      const uint4* ks = (const uint4*)(kbatch + (long)j0 * 64);
      uint4* kd = (uint4*)&Kch[cur][0];
      for (int i = tid; i < 256; i += 128) kd[i] = ks[i];          // 32x64 bf16
      // V^T rows are strided by S; copy 64 rows x 32 elems (4x16B per row).
      for (int i = tid; i < 256; i += 128) {
        const int row = i >> 2, q = i & 3;
        ((uint4*)&Vch[cur][0])[i] =
            ((const uint4*)(vbatch + (long)row * 2048 + j0))[q];
      }
      __syncthreads();
    }
#endif

    // ---- scores S = (Q K^T) * 1/sqrt(64): 16x32 tile from LDS K chunk ----
    v8f sc[2];
#pragma unroll
    for (int nb = 0; nb < 2; ++nb) {
      const bf16* kb = &Kch[cur][(nb * 16) * 64];
      v16bf kb0 = load_frag16(kb + 0, 64, lane);
      v16bf kb1 = load_frag16(kb + 32, 64, lane);
      v8f c;
#pragma unroll
      for (int i = 0; i < 8; ++i) c[i] = 0.f;
      c = __builtin_amdgcn_wmma_f32_16x16x32_bf16(false, qa0, false, kb0,
                                                  (short)0, c, false, false);
      c = __builtin_amdgcn_wmma_f32_16x16x32_bf16(false, qa1, false, kb1,
                                                  (short)0, c, false, false);
#pragma unroll
      for (int i = 0; i < 8; ++i) c[i] *= 0.125f;  // 1/sqrt(d_k)
      sc[nb] = c;
    }

    // ---- online softmax: a row spans the 16 lanes of a half (xor <= 8) ----
    float cmax[8];
#pragma unroll
    for (int i = 0; i < 8; ++i) cmax[i] = fmaxf(sc[0][i], sc[1][i]);
#pragma unroll
    for (int m = 8; m >= 1; m >>= 1)
#pragma unroll
      for (int i = 0; i < 8; ++i)
        cmax[i] = fmaxf(cmax[i], __shfl_xor(cmax[i], m, 32));

    float csum[8];
#pragma unroll
    for (int i = 0; i < 8; ++i) {
      const float mn = fmaxf(rmax[i], cmax[i]);
      const float al = __expf(rmax[i] - mn);
      rmax[i] = mn;
      const float p0 = __expf(sc[0][i] - mn);
      const float p1 = __expf(sc[1][i] - mn);
      // C-layout -> row-major P tile in LDS (same-wave DS ops are in-order)
      pl[(i + 8 * half) * 32 + r] = f2bf(p0);
      pl[(i + 8 * half) * 32 + 16 + r] = f2bf(p1);
      csum[i] = p0 + p1;
      rsum[i] *= al;
#pragma unroll
      for (int t = 0; t < 4; ++t) oacc[t][i] *= al;
    }
#pragma unroll
    for (int m = 8; m >= 1; m >>= 1)
#pragma unroll
      for (int i = 0; i < 8; ++i) csum[i] += __shfl_xor(csum[i], m, 32);
#pragma unroll
    for (int i = 0; i < 8; ++i) rsum[i] += csum[i];

    // ---- O += P(16x32) @ Vchunk(32x64): A-frag from LDS P, B-frags from
    //      LDS V^T chunk (rows = d_k cols, contiguous along keys) ----
    v16bf pa = load_frag16(pl, 32, lane);
#pragma unroll
    for (int t = 0; t < 4; ++t) {
      v16bf vb = load_frag16(&Vch[cur][(t * 16) * 32], 32, lane);
      oacc[t] = __builtin_amdgcn_wmma_f32_16x16x32_bf16(
          false, pa, false, vb, (short)0, oacc[t], false, false);
    }

#if HAVE_TDM
    __builtin_amdgcn_s_wait_tensorcnt(0);  // next-chunk DMA landed
#endif
    __syncthreads();                       // all waves done with buffers
  }

  // ---- normalize and scatter to [B,S,H*64] context ----
#pragma unroll
  for (int i = 0; i < 8; ++i) rsum[i] = 1.f / rsum[i];
#pragma unroll
  for (int t = 0; t < 4; ++t)
#pragma unroll
    for (int i = 0; i < 8; ++i) {
      const int srow = m0 + i + 8 * half;
      const int col = h * 64 + t * 16 + r;
      Ctx[((long)b * 2048 + srow) * 1024 + col] = f2bf(oacc[t][i] * rsum[i]);
    }
}

// ---------------------------------------------------------------------------
extern "C" void kernel_launch(void* const* d_in, const int* in_sizes, int n_in,
                              void* d_out, int out_size, void* d_ws, size_t ws_size,
                              hipStream_t stream) {
  (void)in_sizes; (void)n_in; (void)out_size; (void)ws_size;
  const float* Q  = (const float*)d_in[0];
  const float* K  = (const float*)d_in[1];
  const float* V  = (const float*)d_in[2];
  const float* WQ = (const float*)d_in[3];
  const float* bQ = (const float*)d_in[4];
  const float* WK = (const float*)d_in[5];
  const float* bK = (const float*)d_in[6];
  const float* WV = (const float*)d_in[7];
  const float* bV = (const float*)d_in[8];
  const float* WO = (const float*)d_in[9];
  const float* bO = (const float*)d_in[10];
  float* out = (float*)d_out;

  constexpr long B = 2, S = 2048, D = 1024, H = 16, DK = 64, M = B * S;

  char* base = (char*)d_ws;
  size_t off = 0;
  auto take = [&](long elems) -> bf16* {
    bf16* p = (bf16*)(base + off);
    off += ((size_t)elems * 2 + 255) & ~(size_t)255;
    return p;
  };
  bf16* Qb  = take(M * D);
  bf16* Kb  = take(M * D);
  bf16* Vb  = take(M * D);
  bf16* WQb = take(D * D);
  bf16* WKb = take(DK * D);
  bf16* WVb = take(DK * D);
  bf16* WOb = take(D * D);
  bf16* qp  = take(M * D);       // [B,H,S,DK]
  bf16* kp  = take(B * S * DK);  // [B,S,DK]
  bf16* vT  = take(B * DK * S);  // [B,DK,S]
  bf16* ctx = take(M * D);       // [B,S,D]

  auto cvt = [&](const float* s, bf16* d, long n) {
    long blocks = (n + 255) / 256;
    if (blocks > 4096) blocks = 4096;
    f32_to_bf16_kernel<<<dim3((unsigned)blocks), dim3(256), 0, stream>>>(s, d, n);
  };
  cvt(Q, Qb, M * D);
  cvt(K, Kb, M * D);
  cvt(V, Vb, M * D);
  cvt(WQ, WQb, D * D);
  cvt(WK, WKb, DK * D);
  cvt(WV, WVb, DK * D);
  cvt(WO, WOb, D * D);

  const dim3 blk(256);
  // q = Q @ WQ^T + bQ  -> bf16 [B,H,S,DK]
  gemm_bt_kernel<<<dim3(M / 128, D / 64), blk, 0, stream>>>(
      Qb, WQb, bQ, nullptr, qp, (int)M, (int)D, (int)D, 2);
  // k = K @ WK^T + bK  -> bf16 [B,S,DK]
  gemm_bt_kernel<<<dim3(M / 128, DK / 64), blk, 0, stream>>>(
      Kb, WKb, bK, nullptr, kp, (int)M, (int)DK, (int)D, 1);
  // v = V @ WV^T + bV  -> bf16 [B,DK,S] (transposed)
  gemm_bt_kernel<<<dim3(M / 128, DK / 64), blk, 0, stream>>>(
      Vb, WVb, bV, nullptr, vT, (int)M, (int)DK, (int)D, 3);

  // flash attention: B*H*(S/16) = 4096 waves, 4 waves/block
  mqa_flash_kernel<<<dim3((unsigned)(B * H * (S / 16) / 4)), dim3(128), 0, stream>>>(
      qp, kp, vT, ctx);

  // out = ctx @ WO^T + bO  -> f32
  gemm_bt_kernel<<<dim3(M / 128, D / 64), blk, 0, stream>>>(
      ctx, WOb, bO, out, nullptr, (int)M, (int)D, (int)D, 0);
}
